// Decoder_72086731096094
// MI455X (gfx1250) — matrix operand order repaired and checked by reading, not compiled, not run on hardware
//
#include <hip/hip_runtime.h>
#include <hip/hip_bf16.h>
#include <math.h>

typedef __attribute__((ext_vector_type(16))) _Float16 v16h;
typedef __attribute__((ext_vector_type(8)))  _Float16 v8h;
typedef __attribute__((ext_vector_type(8)))  float    v8f;

#define B_   64
#define T_   8
#define S_   128
#define U_   1024
#define EMB_ 256
#define V_   32000

// ---------------------------------------------------------------------------
// Tiled f16-WMMA GEMM: C[M,N] = act(A[M,K] @ B[K,N] + bias)
// Block tile 64(M) x 128(N), K-step 32. 8 waves: wave (wr=w>>1, wc=w&1)
// computes rows [wr*16,+16) x cols [wc*64,+64) -> 4 v_wmma_f32_16x16x32_f16
// per K step reusing one A fragment.
// A tile kept row-major [m][k]; B tile stored TRANSPOSED [n][k] so that each
// lane's 16 K-contiguous fragment halves are contiguous -> ds_load_b128.
// Requires M%64==0, N%128==0, K%32==0 (true for every GEMM in this model).
// ---------------------------------------------------------------------------
template<bool BIAS, int ACT>
__global__ __launch_bounds__(256) void gemm_wmma(
    const float* __restrict__ A, const float* __restrict__ Bm,
    const float* __restrict__ bias, float* __restrict__ C,
    int M, int N, int K)
{
    __shared__ __attribute__((aligned(16))) _Float16 As[64][40];    // [m][k] +8 pad
    __shared__ __attribute__((aligned(16))) _Float16 BsT[128][40];  // [n][k] +8 pad

    const int tid  = threadIdx.x;
    const int n0   = blockIdx.x * 128;
    const int m0   = blockIdx.y * 64;
    const int lane = tid & 31;
    const int w    = tid >> 5;
    const int wr   = w >> 1;             // 0..3 -> row group
    const int wc   = w & 1;              // 0..1 -> col group (64 cols)
    const int lm   = lane & 15;
    const int lk   = (lane >> 4) << 3;   // A-frag K base: 0 or 8
    const int bk   = (lane >> 4) << 3;   // B-frag K base: 0 or 8

    v8f acc[4] = {{}, {}, {}, {}};

    // A staging: 64x32 = 2048 elems, 8 per thread (coalesced b128 reads)
    const int aIdx = tid * 8;
    const int ar = aIdx >> 5, ac = aIdx & 31;
    // B staging: 32x128 = 4096 elems, 16 per thread. Thread owns column n,
    // walks 16 K values (each global read coalesced across lanes).
    const int bn = tid & 127;
    const int ks = (tid >> 7) << 4;      // 0 or 16

    for (int k0 = 0; k0 < K; k0 += 32) {
        // ---- stage A tile (fp32 -> f16), row-major ----
        {
            const float* src = A + (size_t)(m0 + ar) * K + (k0 + ac);
            if (k0 + 32 < K) __builtin_prefetch(src + 32, 0, 1);
#pragma unroll
            for (int j = 0; j < 8; ++j) As[ar][ac + j] = (_Float16)src[j];
        }
        // ---- stage B tile (fp32 -> f16), TRANSPOSED to [n][k] ----
        {
            const float* src = Bm + (size_t)(k0 + ks) * N + (n0 + bn);
            if (k0 + 32 < K) __builtin_prefetch(src + (size_t)32 * N, 0, 1);
            v8h t0, t1;
#pragma unroll
            for (int j = 0; j < 8; ++j) t0[j] = (_Float16)src[(size_t)j * N];
#pragma unroll
            for (int j = 0; j < 8; ++j) t1[j] = (_Float16)src[(size_t)(8 + j) * N];
            *(v8h*)&BsT[bn][ks]     = t0;
            *(v8h*)&BsT[bn][ks + 8] = t1;
        }
        __syncthreads();

        // ---- A fragment: two contiguous 8-half runs -> ds_load_b128 x2 ----
        v16h a;
        {
            const v8h alo = *(const v8h*)&As[wr * 16 + lm][lk];
            const v8h ahi = *(const v8h*)&As[wr * 16 + lm][lk + 16];
#pragma unroll
            for (int j = 0; j < 8; ++j) { a[j] = alo[j]; a[8 + j] = ahi[j]; }
        }
        // ---- 4 B fragments, 4 WMMAs ----
#pragma unroll
        for (int i = 0; i < 4; ++i) {
            const int col = wc * 64 + i * 16 + lm;
            const v8h blo = *(const v8h*)&BsT[col][bk];
            const v8h bhi = *(const v8h*)&BsT[col][bk + 16];
            v16h bf;
#pragma unroll
            for (int j = 0; j < 8; ++j) { bf[j] = blo[j]; bf[8 + j] = bhi[j]; }
            acc[i] = __builtin_amdgcn_wmma_f32_16x16x32_f16(
                false, a, false, bf, (short)0, acc[i], false, false);
        }
        __syncthreads();
    }

    // C/D layout: VGPR e, lanes 0-15 -> M=e, N=lane; lanes 16-31 -> M=8+e
    const int row0 = m0 + wr * 16 + ((lane >> 4) << 3);
#pragma unroll
    for (int i = 0; i < 4; ++i) {
        const int nc = n0 + wc * 64 + i * 16 + lm;
        const float bv = BIAS ? bias[nc] : 0.f;
#pragma unroll
        for (int e = 0; e < 8; ++e) {
            float v = acc[i][e] + bv;
            if (ACT == 1) v = tanhf(v);
            C[(size_t)(row0 + e) * N + nc] = v;
        }
    }
}

// ---------------------------------------------------------------------------
// Embedding gather: vec[bt, e] = emb[new_tokens[bt], e]   (512 x 256)
// ---------------------------------------------------------------------------
__global__ void embed_gather(const int* __restrict__ tok,
                             const float* __restrict__ emb,
                             float* __restrict__ vec)
{
    int idx = blockIdx.x * blockDim.x + threadIdx.x;      // < 512*256
    int row = idx >> 8, col = idx & 255;
    vec[idx] = emb[(size_t)tok[row] * EMB_ + col];
}

__global__ void copyf(float* __restrict__ dst, const float* __restrict__ src)
{
    int idx = blockIdx.x * blockDim.x + threadIdx.x;
    dst[idx] = src[idx];
}

// ---------------------------------------------------------------------------
// Fused GRU gates for timestep t. 64*1024 threads.
// ---------------------------------------------------------------------------
__global__ void gru_gate(const float* __restrict__ xz_all,   // 512 x 3072
                         const float* __restrict__ hz,       // 64 x 3072
                         float* __restrict__ h,               // 64 x 1024 (in/out)
                         float* __restrict__ rnn,             // 512 x 1024
                         float* __restrict__ state_out,       // 64 x 1024
                         int t)
{
    int idx = blockIdx.x * blockDim.x + threadIdx.x;          // < 65536
    int b = idx >> 10, d = idx & 1023;
    const float* xr_ = xz_all + (size_t)(b * T_ + t) * (3 * U_);
    const float* hr_ = hz + (size_t)b * (3 * U_);
    float xz = xr_[d],        xr = xr_[U_ + d],  xh  = xr_[2 * U_ + d];
    float hzv = hr_[d],       hr = hr_[U_ + d],  hh_ = hr_[2 * U_ + d];
    float z = 1.f / (1.f + __expf(-(xz + hzv)));
    float r = 1.f / (1.f + __expf(-(xr + hr)));
    float hh = tanhf(xh + r * hh_);
    float hn = z * h[idx] + (1.f - z) * hh;
    h[idx] = hn;
    rnn[(size_t)(b * T_ + t) * U_ + d] = hn;
    if (t == T_ - 1) state_out[idx] = hn;
}

// ---------------------------------------------------------------------------
// Fused additive attention: one block per (b,t).
// scores[s] = sum_d tanh(q[d] + k[b,s,d]) * scale[d]  -> mask -> softmax
// context[d] = sum_s p[s] * enc[b,s,d]; writes attn to d_out and
// [context | rnn] into cat buffer for the Wc GEMM.
// ---------------------------------------------------------------------------
__global__ __launch_bounds__(256) void attn_ctx(
    const float* __restrict__ w1q,   // 512 x 1024
    const float* __restrict__ w2k,   // 8192 x 1024
    const float* __restrict__ enc,   // 64 x 128 x 1024
    const unsigned char* __restrict__ mask,  // 64 x 128 (bool)
    const float* __restrict__ scale, // 1024
    const float* __restrict__ rnn,   // 512 x 1024
    float* __restrict__ attn_out,    // 512 x 128
    float* __restrict__ cat)         // 512 x 2048
{
    __shared__ float qs[U_];
    __shared__ float sc[U_];
    __shared__ float scores[S_];

    const int bt = blockIdx.x;       // 0..511
    const int b  = bt >> 3;
    const int tid = threadIdx.x;
    const int w = tid >> 5, lane = tid & 31;

    for (int d = tid; d < U_; d += 256) {
        qs[d] = w1q[(size_t)bt * U_ + d];
        sc[d] = scale[d];
    }
    __syncthreads();

    for (int s = w; s < S_; s += 8) {
        const float* kp = w2k + (size_t)(b * S_ + s) * U_;
        float acc = 0.f;
        for (int d = lane; d < U_; d += 32)
            acc += tanhf(qs[d] + kp[d]) * sc[d];
#pragma unroll
        for (int off = 16; off; off >>= 1) acc += __shfl_xor(acc, off, 32);
        if (lane == 0)
            scores[s] = mask[b * S_ + s] ? acc : -1.0e9f;
    }
    __syncthreads();

    // softmax over 128 (all threads redundantly reduce from LDS)
    float mx = -1.0e30f;
    for (int s = 0; s < S_; ++s) mx = fmaxf(mx, scores[s]);
    float sum = 0.f;
    for (int s = 0; s < S_; ++s) sum += __expf(scores[s] - mx);
    float inv = 1.f / sum;
    __syncthreads();
    if (tid < S_) {
        float p = __expf(scores[tid] - mx) * inv;
        scores[tid] = p;
        attn_out[(size_t)bt * S_ + tid] = p;
    }
    __syncthreads();

    // context + concat
    for (int d = tid; d < U_; d += 256) {
        const float* ep = enc + (size_t)b * S_ * U_ + d;
        float acc = 0.f;
        for (int s = 0; s < S_; ++s) acc += scores[s] * ep[(size_t)s * U_];
        cat[(size_t)bt * 2 * U_ + d]      = acc;
        cat[(size_t)bt * 2 * U_ + U_ + d] = rnn[(size_t)bt * U_ + d];
    }
}

// ---------------------------------------------------------------------------
extern "C" void kernel_launch(void* const* d_in, const int* in_sizes, int n_in,
                              void* d_out, int out_size, void* d_ws, size_t ws_size,
                              hipStream_t stream)
{
    (void)in_sizes; (void)n_in; (void)out_size; (void)ws_size;

    const int*   tok   = (const int*)  d_in[0];
    const float* enc   = (const float*)d_in[1];
    const unsigned char* mask = (const unsigned char*)d_in[2];
    const float* state = (const float*)d_in[3];
    const float* emb   = (const float*)d_in[4];
    const float* Wx    = (const float*)d_in[5];
    const float* bx    = (const float*)d_in[6];
    const float* Wh    = (const float*)d_in[7];
    const float* bh    = (const float*)d_in[8];
    const float* W1    = (const float*)d_in[9];
    const float* W2    = (const float*)d_in[10];
    const float* ascal = (const float*)d_in[11];
    const float* Wc    = (const float*)d_in[12];
    const float* fcW   = (const float*)d_in[13];
    const float* fcb   = (const float*)d_in[14];

    float* logits    = (float*)d_out;                       // 512 x 32000
    float* attn_out  = logits + (size_t)B_ * T_ * V_;       // 512 x 128
    float* state_out = attn_out + (size_t)B_ * T_ * S_;     // 64 x 1024

    float* ws  = (float*)d_ws;
    float* vec = ws;                      // 512 x 256
    float* xz  = vec + 512 * 256;         // 512 x 3072
    float* hz  = xz  + 512 * 3072;        // 64 x 3072
    float* h   = hz  + 64 * 3072;         // 64 x 1024
    float* rnn = h   + 64 * 1024;         // 512 x 1024
    float* w1q = rnn + 512 * 1024;        // 512 x 1024
    float* w2k = w1q + 512 * 1024;        // 8192 x 1024
    float* cat = w2k + 8192 * 1024;       // 512 x 2048
    float* av  = cat + 512 * 2048;        // 512 x 1024

    // 1. embedding gather + h0
    embed_gather<<<512, 256, 0, stream>>>(tok, emb, vec);
    copyf<<<256, 256, 0, stream>>>(h, state);

    // 2. xz_all = vec @ Wx + bx   (512 x 3072, K=256)
    gemm_wmma<true, 0><<<dim3(3072 / 128, 512 / 64), 256, 0, stream>>>(
        vec, Wx, bx, xz, 512, 3072, 256);

    // 3. GRU scan over T=8
    for (int t = 0; t < T_; ++t) {
        gemm_wmma<true, 0><<<dim3(3072 / 128, 1), 256, 0, stream>>>(
            h, Wh, bh, hz, 64, 3072, 1024);
        gru_gate<<<256, 256, 0, stream>>>(xz, hz, h, rnn, state_out, t);
    }

    // 4. attention projections
    gemm_wmma<false, 0><<<dim3(1024 / 128, 512 / 64), 256, 0, stream>>>(
        rnn, W1, nullptr, w1q, 512, 1024, 1024);
    gemm_wmma<false, 0><<<dim3(1024 / 128, 8192 / 64), 256, 0, stream>>>(
        enc, W2, nullptr, w2k, 8192, 1024, 1024);

    // 5. scores -> softmax -> context, concat [context|rnn]
    attn_ctx<<<512, 256, 0, stream>>>(w1q, w2k, enc, mask, ascal, rnn,
                                      attn_out, cat);

    // 6. attention_vector = tanh(cat @ Wc)   (512 x 1024, K=2048)
    gemm_wmma<false, 1><<<dim3(1024 / 128, 512 / 64), 256, 0, stream>>>(
        cat, Wc, nullptr, av, 512, 1024, 2048);

    // 7. logits = av @ fc_W + fc_b   (512 x 32000, K=1024)
    gemm_wmma<true, 0><<<dim3(32000 / 128, 512 / 64), 256, 0, stream>>>(
        av, fcW, fcb, logits, 512, 32000, 1024);
}
